// Model_21689584845411
// MI455X (gfx1250) — matrix operand order repaired and checked by reading, not compiled, not run on hardware
//
#include <hip/hip_runtime.h>

// ---------------------------------------------------------------------------
// Model constants (from reference): B=4096, S=T=16, V=128, E=256, H=512
// ---------------------------------------------------------------------------
static constexpr int kB  = 4096;
static constexpr int kS  = 16;
static constexpr int kT  = 16;
static constexpr int kV  = 128;
static constexpr int kE  = 256;
static constexpr int kH  = 512;
static constexpr int kH3 = 3 * kH;   // 1536

typedef __attribute__((ext_vector_type(16))) __bf16 v16bf;
typedef __attribute__((ext_vector_type(8)))  __bf16 v8bf;
typedef __attribute__((ext_vector_type(8)))  float  v8f;

__device__ __forceinline__ __bf16 f2bf(float f) {
    unsigned u = __builtin_bit_cast(unsigned, f);
    unsigned r = (u + 0x7FFFu + ((u >> 16) & 1u)) >> 16;  // round-to-nearest-even
    return __builtin_bit_cast(__bf16, (unsigned short)r);
}

// A-fragment (16x32 bf16, MxK): per CDNA5 ISA, lane group holds K in two
// contiguous runs of 8 (k0..k0+7 and k0+16..k0+23), k0 = (lane>=16)*8.
__device__ __forceinline__ v16bf load_a_frag(const __bf16* p) {
    v8bf lo = *(const v8bf*)(p);
    v8bf hi = *(const v8bf*)(p + 16);
    return __builtin_shufflevector(lo, hi, 0, 1, 2, 3, 4, 5, 6, 7,
                                           8, 9, 10, 11, 12, 13, 14, 15);
}

// ---------------------------------------------------------------------------
// WMMA GEMM:  C[M,N] = A[M,K](bf16,row-major) * BT[N,K](bf16,row-major)^T + bias
// Block tile 128(M) x 32(N), 8 wave32 (one 16x32 wave tile each, shared A frag).
// The 32xK B tile is staged once per block into LDS with async-to-LDS copies
// (ASYNCcnt), then consumed via ds_load; A streams from global with prefetch.
// grid = (N/32, M/128), block = 256 threads.
// ---------------------------------------------------------------------------
__global__ __launch_bounds__(256)
void gemm_bf16_wmma(const __bf16* __restrict__ A, int lda, int K,
                    const __bf16* __restrict__ BT,
                    const float* __restrict__ bias,
                    float* __restrict__ C, int ldc, int coff)
{
    // 32 rows (block's N columns) x K (<=512) bf16 = 32 KB of the 320 KB WGP LDS
    __shared__ __align__(32) __bf16 sB[32 * 512];

    const int tid  = threadIdx.x;
    const int lane = threadIdx.x & 31;
    const int wave = threadIdx.x >> 5;
    const int m0   = blockIdx.y * 128 + wave * 16;
    const int n0   = blockIdx.x * 32;
    const int half = lane >> 4;      // lane group 0/1
    const int rl   = lane & 15;

    // ---- stage B tile into LDS: 256 lanes x 16B async copies ----
    {
        const int iters = (32 * K) / (256 * 8);   // K=256 -> 4, K=512 -> 8
        for (int it = 0; it < iters; ++it) {
            int eo  = (it * 256 + tid) * 8;       // bf16 element offset in tile
            int row = eo / K;                     // local n (0..31)
            int col = eo - row * K;
            unsigned long long ga =
                (unsigned long long)(uintptr_t)(BT + (size_t)(n0 + row) * K + col);
            // Low 32 bits of a generic LDS address are the wave-relative LDS
            // byte address; taking it from the
            // sB pointer also makes sB "captured" so the asm's memory clobber
            // aliases it and the ds reads below cannot be folded away.
            unsigned ldsAddr = (unsigned)(uintptr_t)(sB + eo);
            asm volatile("global_load_async_to_lds_b128 %0, %1, off"
                         :: "v"(ldsAddr), "v"(ga) : "memory");
        }
        asm volatile("s_wait_asynccnt 0x0" ::: "memory");
    }
    __syncthreads();

    v8f acc0, acc1;
    const float bv0 = bias ? bias[n0 + rl]      : 0.0f;
    const float bv1 = bias ? bias[n0 + 16 + rl] : 0.0f;
#pragma unroll
    for (int i = 0; i < 8; ++i) { acc0[i] = bv0; acc1[i] = bv1; }

    // A: row m = m0 + rl, K-runs start at half*8 (then +16 inside frag)
    const __bf16* ap  = A + (size_t)(m0 + rl) * lda + half * 8;
    // B fragments from LDS: row n-local = rl (+16 for second tile), K-runs at half*8 (then +16)
    const __bf16* sb0 = sB + (size_t)rl * K + half * 8;
    const __bf16* sb1 = sB + (size_t)(16 + rl) * K + half * 8;

    for (int kb = 0; kb < K; kb += 32) {
        __builtin_prefetch(ap + kb + 128, 0, 3);  // near-temporal prefetch of A stream
        v16bf af = load_a_frag(ap + kb);
        v16bf b0 = load_a_frag(sb0 + kb);
        v16bf b1 = load_a_frag(sb1 + kb);
        acc0 = __builtin_amdgcn_wmma_f32_16x16x32_bf16(false, af, false, b0,
                                                       (short)0, acc0, false, false);
        acc1 = __builtin_amdgcn_wmma_f32_16x16x32_bf16(false, af, false, b1,
                                                       (short)0, acc1, false, false);
    }

    // C/D layout: VGPR v -> row m0 + v + half*8, col n0 + rl (+16 for acc1)
    float* cp = C + (size_t)coff + (size_t)(m0 + half * 8) * ldc + n0 + rl;
#pragma unroll
    for (int v = 0; v < 8; ++v) {
        cp[(size_t)v * ldc]      = acc0[v];
        cp[(size_t)v * ldc + 16] = acc1[v];
    }
}

// ---------------------------------------------------------------------------
// One-time prep kernels
// ---------------------------------------------------------------------------
// W[K,N] fp32 -> WT[N,K] bf16  (coalesced read of W)
__global__ void transpose_w_bf(const float* __restrict__ W, __bf16* __restrict__ WT,
                               int K, int N)
{
    int i = blockIdx.x * blockDim.x + threadIdx.x;
    if (i >= K * N) return;
    int n = i % N, k = i / N;
    WT[(size_t)n * K + k] = f2bf(W[i]);
}

// idx[B,Slen] -> out[Slen,B,E] bf16 gathered from emb[V,E]
__global__ void gather_emb(const int* __restrict__ idx, const float* __restrict__ emb,
                           __bf16* __restrict__ out, int Blen, int Slen)
{
    int i = blockIdx.x * blockDim.x + threadIdx.x;
    if (i >= Slen * Blen * kE) return;
    int e = i % kE;
    int t = i / kE;
    int b = t % Blen;
    int s = t / Blen;
    out[i] = f2bf(emb[(size_t)idx[b * Slen + s] * kE + e]);
}

// broadcast emb[row,:] to out[B,E] bf16 (BOW input)
__global__ void bcast_row_emb(const float* __restrict__ emb, int row,
                              __bf16* __restrict__ out)
{
    int i = blockIdx.x * blockDim.x + threadIdx.x;
    if (i >= kB * kE) return;
    out[i] = f2bf(emb[(size_t)row * kE + (i % kE)]);
}

__global__ void zero_h(float* __restrict__ h, __bf16* __restrict__ hbf)
{
    int i = blockIdx.x * blockDim.x + threadIdx.x;
    if (i >= kB * kH) return;
    h[i] = 0.0f;
    hbf[i] = f2bf(0.0f);
}

// merge_mode='sum': hf += hb, refresh bf16 shadow
__global__ void merge_sum(float* __restrict__ hf, const float* __restrict__ hb,
                          __bf16* __restrict__ hbf)
{
    int i = blockIdx.x * blockDim.x + threadIdx.x;
    if (i >= kB * kH) return;
    float s = hf[i] + hb[i];
    hf[i] = s;
    hbf[i] = f2bf(s);
}

// Keras GRU gates (reset_after=True, order [z,r,h]):
//   z = sig(xz+hz); r = sig(xr+hr); hc = tanh(xh + r*hh); h = z*h + (1-z)*hc
__global__ void gru_gate(const float* __restrict__ XP, const float* __restrict__ HP,
                         float* __restrict__ h, __bf16* __restrict__ hbf)
{
    int i = blockIdx.x * blockDim.x + threadIdx.x;
    if (i >= kB * kH) return;
    int b = i / kH, j = i % kH;
    const float* xp = XP + (size_t)b * kH3;
    const float* hp = HP + (size_t)b * kH3;
    float z  = 1.0f / (1.0f + __expf(-(xp[j] + hp[j])));
    float r  = 1.0f / (1.0f + __expf(-(xp[kH + j] + hp[kH + j])));
    float hc = tanhf(xp[2 * kH + j] + r * hp[2 * kH + j]);
    float hn = z * h[i] + (1.0f - z) * hc;
    h[i] = hn;
    hbf[i] = f2bf(hn);
}

// ---------------------------------------------------------------------------
// Host orchestration
// ---------------------------------------------------------------------------
extern "C" void kernel_launch(void* const* d_in, const int* in_sizes, int n_in,
                              void* d_out, int out_size, void* d_ws, size_t ws_size,
                              hipStream_t stream)
{
    (void)in_sizes; (void)n_in; (void)out_size; (void)ws_size;

    const int*   sources    = (const int*)  d_in[0];
    const int*   targets    = (const int*)  d_in[1];
    const float* src_emb    = (const float*)d_in[2];
    const float* enc_fwd_Wx = (const float*)d_in[3];
    const float* enc_fwd_Wh = (const float*)d_in[4];
    const float* enc_fwd_bx = (const float*)d_in[5];
    const float* enc_fwd_bh = (const float*)d_in[6];
    const float* enc_bwd_Wx = (const float*)d_in[7];
    const float* enc_bwd_Wh = (const float*)d_in[8];
    const float* enc_bwd_bx = (const float*)d_in[9];
    const float* enc_bwd_bh = (const float*)d_in[10];
    const float* tgt_emb    = (const float*)d_in[11];
    const float* dec_Wx     = (const float*)d_in[12];
    const float* dec_Wh     = (const float*)d_in[13];
    const float* dec_bx     = (const float*)d_in[14];
    const float* dec_bh     = (const float*)d_in[15];
    const float* out_W      = (const float*)d_in[16];
    const float* out_b      = (const float*)d_in[17];
    float*       logits     = (float*)d_out;   // [B, T, V] fp32

    // ---- carve workspace ----
    char* wp = (char*)d_ws;
    auto carve = [&](size_t bytes) -> char* {
        char* q = wp;
        wp += (bytes + 255) & ~(size_t)255;
        return q;
    };
    __bf16* wfxT  = (__bf16*)carve((size_t)kH3 * kE * 2);   // [3H, E]
    __bf16* wfhT  = (__bf16*)carve((size_t)kH3 * kH * 2);   // [3H, H]
    __bf16* wbxT  = (__bf16*)carve((size_t)kH3 * kE * 2);
    __bf16* wbhT  = (__bf16*)carve((size_t)kH3 * kH * 2);
    __bf16* wdxT  = (__bf16*)carve((size_t)kH3 * kE * 2);
    __bf16* wdhT  = (__bf16*)carve((size_t)kH3 * kH * 2);
    __bf16* woutT = (__bf16*)carve((size_t)kV  * kH * 2);   // [V, H]
    __bf16* srcbf = (__bf16*)carve((size_t)kS * kB * kE * 2);  // [S,B,E]
    __bf16* tgtbf = (__bf16*)carve((size_t)kT * kB * kE * 2);  // [T,B,E]
    __bf16* bowbf = (__bf16*)carve((size_t)kB * kE * 2);       // [B,E]
    float*  hfwd  = (float*) carve((size_t)kB * kH * 4);
    float*  hbwd  = (float*) carve((size_t)kB * kH * 4);
    __bf16* hbf   = (__bf16*)carve((size_t)kB * kH * 2);       // bf16 shadow of "current h"
    float*  xp    = (float*) carve((size_t)kB * kH3 * 4);
    float*  hp    = (float*) carve((size_t)kB * kH3 * 4);

    const int TPB = 256;
    auto blocks = [](long long n) { return (unsigned)((n + 255) / 256); };

    auto gemm = [&](const __bf16* A, int lda, int K, const __bf16* BT,
                    const float* bias, float* C, int ldc, int coff, int M, int N) {
        dim3 grid((unsigned)(N / 32), (unsigned)(M / 128));
        gemm_bf16_wmma<<<grid, TPB, 0, stream>>>(A, lda, K, BT, bias, C, ldc, coff);
    };

    // ---- one-time prep: weights -> bf16 [N,K]; embeddings -> bf16 ----
    transpose_w_bf<<<blocks((long long)kE * kH3), TPB, 0, stream>>>(enc_fwd_Wx, wfxT, kE, kH3);
    transpose_w_bf<<<blocks((long long)kH * kH3), TPB, 0, stream>>>(enc_fwd_Wh, wfhT, kH, kH3);
    transpose_w_bf<<<blocks((long long)kE * kH3), TPB, 0, stream>>>(enc_bwd_Wx, wbxT, kE, kH3);
    transpose_w_bf<<<blocks((long long)kH * kH3), TPB, 0, stream>>>(enc_bwd_Wh, wbhT, kH, kH3);
    transpose_w_bf<<<blocks((long long)kE * kH3), TPB, 0, stream>>>(dec_Wx,     wdxT, kE, kH3);
    transpose_w_bf<<<blocks((long long)kH * kH3), TPB, 0, stream>>>(dec_Wh,     wdhT, kH, kH3);
    transpose_w_bf<<<blocks((long long)kH * kV ), TPB, 0, stream>>>(out_W,      woutT, kH, kV);

    gather_emb<<<blocks((long long)kS * kB * kE), TPB, 0, stream>>>(sources, src_emb, srcbf, kB, kS);
    gather_emb<<<blocks((long long)kT * kB * kE), TPB, 0, stream>>>(targets, tgt_emb, tgtbf, kB, kT);
    bcast_row_emb<<<blocks((long long)kB * kE), TPB, 0, stream>>>(tgt_emb, /*BOW=*/1, bowbf);

    const unsigned hblk = blocks((long long)kB * kH);

    // ---- encoder forward GRU ----
    zero_h<<<hblk, TPB, 0, stream>>>(hfwd, hbf);
    for (int s = 0; s < kS; ++s) {
        gemm(srcbf + (size_t)s * kB * kE, kE, kE, wfxT, enc_fwd_bx, xp, kH3, 0, kB, kH3);
        gemm(hbf, kH, kH, wfhT, enc_fwd_bh, hp, kH3, 0, kB, kH3);
        gru_gate<<<hblk, TPB, 0, stream>>>(xp, hp, hfwd, hbf);
    }

    // ---- encoder backward GRU (reversed sequence) ----
    zero_h<<<hblk, TPB, 0, stream>>>(hbwd, hbf);
    for (int s = 0; s < kS; ++s) {
        int sr = kS - 1 - s;
        gemm(srcbf + (size_t)sr * kB * kE, kE, kE, wbxT, enc_bwd_bx, xp, kH3, 0, kB, kH3);
        gemm(hbf, kH, kH, wbhT, enc_bwd_bh, hp, kH3, 0, kB, kH3);
        gru_gate<<<hblk, TPB, 0, stream>>>(xp, hp, hbwd, hbf);
    }

    // ---- merge (sum) -> decoder initial state in hfwd/hbf ----
    merge_sum<<<hblk, TPB, 0, stream>>>(hfwd, hbwd, hbf);

    // ---- decoder: teacher-forced GRU + dense logits per step ----
    for (int t = 0; t < kT; ++t) {
        const __bf16* xin = (t == 0) ? bowbf : (tgtbf + (size_t)(t - 1) * kB * kE);
        gemm(xin, kE, kE, wdxT, dec_bx, xp, kH3, 0, kB, kH3);
        gemm(hbf, kH, kH, wdhT, dec_bh, hp, kH3, 0, kB, kH3);
        gru_gate<<<hblk, TPB, 0, stream>>>(xp, hp, hfwd, hbf);
        // logits[:, t, :] = h @ out_W + out_b   (ldc = T*V, offset t*V)
        gemm(hbf, kH, kH, woutT, out_b, logits, kT * kV, t * kV, kB, kV);
    }
}
